// MRIMambaFormer_66288525247130
// MI455X (gfx1250) — compile-verified, matrix operand order checked by reading, not run on hardware
//
#include <hip/hip_runtime.h>
#include <hip/hip_bf16.h>
#include <math.h>

// ---------------------------------------------------------------------------
// MRIMambaFormer on MI455X (gfx1250, wave32, WMMA + TDM).
// Dense projections: v_wmma_f32_16x16x32_bf16 with the shared A-tile staged
// into LDS by the Tensor Data Mover (double-buffered, TENSORcnt-tracked),
// B operands streamed from global. Mamba selective scans are register-
// resident VALU kernels with B/C vectors staged through LDS. Cross-attention
// with a single KV token degenerates to a broadcast vector.
// ---------------------------------------------------------------------------

#define D_MODEL  512
#define D_INNER  1024
#define D_STATE  16
#define D_CONV   4
#define DT_RANK  32
#define MRI_DIM  1024
#define CLIN_DIM 64
#define N_SLICES 2000
#define CHUNK    500
#define N_LAYERS 3
#define N_HEADS  8
#define HEAD_DIM 64

typedef __bf16  v16bf __attribute__((ext_vector_type(16)));
typedef float   v8f   __attribute__((ext_vector_type(8)));

#define ACT_NONE     0
#define ACT_SOFTPLUS 1
#define ACT_GELU     2

#ifndef __has_builtin
#define __has_builtin(x) 0
#endif
#if __has_builtin(__builtin_amdgcn_tensor_load_to_lds) && \
    __has_builtin(__builtin_amdgcn_s_wait_tensorcnt)
#define HAVE_TDM 1
#else
#define HAVE_TDM 0
#endif

typedef unsigned int tdm_u32x4 __attribute__((ext_vector_type(4)));
typedef int          tdm_i32x4 __attribute__((ext_vector_type(4)));
typedef int          tdm_i32x8 __attribute__((ext_vector_type(8)));

__device__ __forceinline__ float act_apply(float v, int act) {
    if (act == ACT_SOFTPLUS) {
        return (v > 20.f) ? v : log1pf(__expf(v));
    } else if (act == ACT_GELU) {
        return 0.5f * v * (1.f + erff(v * 0.70710678118654752f));
    }
    return v;
}

// LDS row stride for the staged A tile: 32 data dwords + 2 pad dwords
// (TDM pad_interval = 32 dwords, pad_amount = 2 dwords) to spread banks.
#define A_ROW_STRIDE 34

#if HAVE_TDM
// ---------------------------------------------------------------------------
// Issue a TDM 2D tile load: 16 rows x 32 f32, row stride = lda elements,
// from global (A + m0*lda + k0) into LDS at byte offset lds_off.
// tensor_dim0 = K and tensor_dim1 = M-m0 give hardware OOB zero-fill on the
// K tail / M tail. Descriptor per CDNA5 ISA ch.8 (D# groups 0/1; 2D tensor,
// groups 2/3 unused).
// ---------------------------------------------------------------------------
__device__ __forceinline__ void tdm_issue_tile(const float* A, int lda, int M,
                                               int K, int m0, int k0,
                                               unsigned lds_off)
{
    unsigned long long ga =
        (unsigned long long)(size_t)(A + (size_t)m0 * lda + k0);
    tdm_u32x4 g0;
    g0.x = 1u;                                            // count=1, user D#
    g0.y = lds_off;                                       // LDS byte address
    g0.z = (unsigned)ga;                                  // global_addr[31:0]
    g0.w = (unsigned)((ga >> 32) & 0x01ffffffu) | (2u << 30); // addr[56:32]|type=2
    const unsigned rows = (unsigned)(M - m0);
    tdm_i32x8 g1;
    // data_size=4B (2<<16) | pad_enable (1<<20) | pad_interval=32dw (4<<22)
    // | pad_amount=2dw (1<<25); workgroup_mask=0 (not in a cluster)
    g1[0] = (int)((2u << 16) | (1u << 20) | (4u << 22) | (1u << 25));
    g1[1] = (int)(((unsigned)K & 0xffffu) << 16);         // tensor_dim0 lo16
    g1[2] = (int)((((unsigned)K >> 16) & 0xffffu) | ((rows & 0xffffu) << 16));
    g1[3] = (int)(((rows >> 16) & 0xffffu) | (32u << 16)); // tile_dim0 = 32
    g1[4] = 16;                                            // tile_dim1 = 16
    g1[5] = lda;                                           // dim0_stride lo32
    g1[6] = 0;
    g1[7] = 0;
#if __clang_major__ >= 23
    __builtin_amdgcn_tensor_load_to_lds(g0, g1, (tdm_i32x4)0, (tdm_i32x4)0,
                                        (tdm_i32x8)0, 0);
#else
    __builtin_amdgcn_tensor_load_to_lds(g0, g1, (tdm_i32x4)0, (tdm_i32x4)0, 0);
#endif
}
#endif // HAVE_TDM

// ---------------------------------------------------------------------------
// Generic WMMA GEMM: C[M,N] = act( scale * (A[M,K] * B[N,K]^T) + bias[N] )
// A row-major (lda), B row-major [N,K] (ldb) — the "x @ W.T" layout.
// Block = 256 threads = 8 waves; the block's 16 A rows are staged via TDM
// into a double-buffered LDS tile shared by all waves; each wave computes a
// 16x64 output strip (A fragment reused across 4 WMMAs).
// Fragment layout per CDNA5 ISA 7.12.2 (16-bit A 16x32): lanes 0-15 carry
// K = {0..7, 16..23}, lanes 16-31 carry K = {8..15, 24..31}.
// ---------------------------------------------------------------------------
__global__ void __launch_bounds__(256)
gemm_bf16_wmma(const float* __restrict__ A, int lda,
               const float* __restrict__ B, int ldb,
               float* __restrict__ C, int ldc,
               const float* __restrict__ bias,
               int M, int N, int K, float scale, int act)
{
    __shared__ float Atile[2][16 * A_ROW_STRIDE];

    const int tid     = threadIdx.x;
    const int wave    = tid >> 5;
    const int lane    = tid & 31;
    const int halfsel = lane >> 4;          // 0: K-low half, 1: K-high half
    const int r       = lane & 15;
    const int kb      = halfsel * 8;
    const int m0      = blockIdx.y * 16;
    const int n0base  = (blockIdx.x * 8 + wave) * 64;

    // Clamped per-tile B row pointers (inactive tiles read row N-1, results
    // are discarded at the store). Keeps the K loop branch-free and keeps
    // every wave alive for the block barriers.
    const float* Bp[4];
    #pragma unroll
    for (int t = 0; t < 4; ++t) {
        int brow = n0base + t * 16 + r;
        if (brow >= N) brow = N - 1;
        Bp[t] = B + (size_t)brow * ldb;
    }

    v8f acc[4];
    #pragma unroll
    for (int t = 0; t < 4; ++t)
        #pragma unroll
        for (int i = 0; i < 8; ++i) acc[t][i] = 0.f;

    const int nk = (K + 31) >> 5;

#if HAVE_TDM
    if (tid == 0)
        tdm_issue_tile(A, lda, M, K, m0, 0,
                       (unsigned)(size_t)(void*)&Atile[0][0]);
#endif

    for (int kt = 0; kt < nk; ++kt) {
        const int k0  = kt << 5;
        const int cur = kt & 1;
#if HAVE_TDM
        if (tid == 0)
            __builtin_amdgcn_s_wait_tensorcnt((short)0);  // tile kt landed
        __syncthreads();
        if (tid == 0 && kt + 1 < nk)                      // prefetch kt+1
            tdm_issue_tile(A, lda, M, K, m0, k0 + 32,
                           (unsigned)(size_t)(void*)&Atile[cur ^ 1][0]);
#else
        __syncthreads();
        for (int e = tid; e < 512; e += 256) {
            const int rr = e >> 5, cc = e & 31;
            const int kk = k0 + cc, mm = m0 + rr;
            Atile[cur][rr * A_ROW_STRIDE + cc] =
                (kk < K && mm < M) ? A[(size_t)mm * lda + kk] : 0.f;
        }
        __syncthreads();
#endif
        // A fragment from LDS (tails are zero-filled by TDM OOB / loader)
        const float* __restrict__ Ar = &Atile[cur][r * A_ROW_STRIDE];
        v16bf a;
        #pragma unroll
        for (int j = 0; j < 8; ++j) {
            a[j]     = (__bf16)Ar[kb + j];
            a[8 + j] = (__bf16)Ar[16 + kb + j];
        }

        const bool tail = (k0 + 32 > K);
        #pragma unroll
        for (int t = 0; t < 4; ++t) {
            v16bf b;
            if (!tail) {
                __builtin_prefetch(Bp[t] + k0 + 32, 0, 0);
                #pragma unroll
                for (int j = 0; j < 8; ++j) {
                    b[j]     = (__bf16)Bp[t][k0 + kb + j];
                    b[8 + j] = (__bf16)Bp[t][k0 + 16 + kb + j];
                }
            } else {
                #pragma unroll
                for (int j = 0; j < 8; ++j) {
                    int k1 = k0 + kb + j, k2 = k0 + 16 + kb + j;
                    b[j]     = (__bf16)((k1 < K) ? Bp[t][k1] : 0.f);
                    b[8 + j] = (__bf16)((k2 < K) ? Bp[t][k2] : 0.f);
                }
            }
            acc[t] = __builtin_amdgcn_wmma_f32_16x16x32_bf16(
                false, a, false, b, (short)0, acc[t], false, false);
        }
    }

    // C/D layout: VGPR i -> M = m0 + i + 8*halfsel, N = n0 + (lane & 15)
    #pragma unroll
    for (int t = 0; t < 4; ++t) {
        const int col = n0base + t * 16 + r;
        if (col >= N) continue;
        const float bv = bias ? bias[col] : 0.f;
        #pragma unroll
        for (int i = 0; i < 8; ++i) {
            const int row = m0 + i + halfsel * 8;
            if (row < M)
                C[(size_t)row * ldc + col] = act_apply(acc[t][i] * scale + bv, act);
        }
    }
}

// ---------------------------------------------------------------------------
// Row LayerNorm with optional residual (res_stride=0 broadcasts one vector)
// and optional ReLU.  One 256-thread block per row.
// ---------------------------------------------------------------------------
__global__ void __launch_bounds__(256)
ln_kernel(const float* __restrict__ x, const float* __restrict__ res,
          int res_stride, const float* __restrict__ g,
          const float* __restrict__ b, float* __restrict__ out,
          int D, int relu_flag)
{
    const int row = blockIdx.x;
    const int t   = threadIdx.x;
    const float* xr = x + (size_t)row * D;
    const float* rr = res ? res + (size_t)row * res_stride : nullptr;

    __shared__ float rs[256], rs2[256];
    float s = 0.f, s2 = 0.f;
    for (int i = t; i < D; i += 256) {
        float v = xr[i] + (rr ? rr[i] : 0.f);
        s += v; s2 += v * v;
    }
    rs[t] = s; rs2[t] = s2; __syncthreads();
    for (int st = 128; st > 0; st >>= 1) {
        if (t < st) { rs[t] += rs[t + st]; rs2[t] += rs2[t + st]; }
        __syncthreads();
    }
    const float mean = rs[0] / D;
    const float var  = rs2[0] / D - mean * mean;
    const float inv  = rsqrtf(var + 1e-5f);
    for (int i = t; i < D; i += 256) {
        float v = xr[i] + (rr ? rr[i] : 0.f);
        float o = (v - mean) * inv * g[i] + b[i];
        if (relu_flag) o = fmaxf(o, 0.f);
        out[(size_t)row * D + i] = o;
    }
}

// ---------------------------------------------------------------------------
// Causal depthwise conv (width 4, per-chunk) + SiLU, reading the x half of xz.
// ---------------------------------------------------------------------------
__global__ void __launch_bounds__(256)
conv_silu_kernel(const float* __restrict__ xz, const float* __restrict__ w,
                 const float* __restrict__ cb, float* __restrict__ xc, int Lc)
{
    const int idx = blockIdx.x * 256 + threadIdx.x;
    if (idx >= N_SLICES * D_INNER) return;
    const int d   = idx & (D_INNER - 1);
    const int row = idx >> 10;
    const int l   = row % Lc;
    float acc = cb[d];
    #pragma unroll
    for (int k = 0; k < D_CONV; ++k) {
        const int lk = l + k - (D_CONV - 1);
        if (lk >= 0)
            acc += w[d * D_CONV + k] * xz[(size_t)(row + k - (D_CONV - 1)) * (2 * D_INNER) + d];
    }
    xc[idx] = acc / (1.f + __expf(-acc));       // silu
}

// ---------------------------------------------------------------------------
// Selective scan: one thread owns one channel d (16-state recurrence in
// registers), B/C vectors for the timestep staged through LDS.
// grid.x = numChunks*4, 256 threads (4 blocks cover the 1024 channels).
// y = (sum_s h_s * C_s) + D*xc, then gated by silu(z).
// ---------------------------------------------------------------------------
__global__ void __launch_bounds__(256)
mamba_scan_kernel(const float* __restrict__ dt, const float* __restrict__ xc,
                  const float* __restrict__ proj, const float* __restrict__ xz,
                  const float* __restrict__ A_log, const float* __restrict__ Dp,
                  float* __restrict__ y, int Lc)
{
    const int cb = blockIdx.x >> 2;
    const int d  = (blockIdx.x & 3) * 256 + threadIdx.x;
    float A[D_STATE], h[D_STATE];
    #pragma unroll
    for (int s = 0; s < D_STATE; ++s) {
        A[s] = -__expf(A_log[d * D_STATE + s]);
        h[s] = 0.f;
    }
    const float Dv = Dp[d];
    __shared__ float BC[2 * D_STATE];

    for (int l = 0; l < Lc; ++l) {
        const size_t row = (size_t)cb * Lc + l;
        __syncthreads();
        if (threadIdx.x < 2 * D_STATE)
            BC[threadIdx.x] = proj[row * (DT_RANK + 2 * D_STATE) + DT_RANK + threadIdx.x];
        __syncthreads();
        const float dtv = dt[row * D_INNER + d];
        const float xv  = xc[row * D_INNER + d];
        float acc = 0.f;
        #pragma unroll
        for (int s = 0; s < D_STATE; ++s) {
            h[s] = __expf(dtv * A[s]) * h[s] + dtv * BC[s] * xv;
            acc += h[s] * BC[D_STATE + s];
        }
        float yv = acc + Dv * xv;
        const float zv = xz[row * (2 * D_INNER) + D_INNER + d];
        yv *= zv / (1.f + __expf(-zv));          // * silu(z)
        y[row * D_INNER + d] = yv;
    }
}

// ---------------------------------------------------------------------------
// Row softmax over L columns (scores pre-scaled by 1/sqrt(hd) in the GEMM).
// ---------------------------------------------------------------------------
__global__ void __launch_bounds__(256)
softmax_rows_kernel(float* __restrict__ p, int L)
{
    const int row = blockIdx.x;
    const int t   = threadIdx.x;
    float* pr = p + (size_t)row * L;
    __shared__ float red[256];

    float mx = -3.4e38f;
    for (int i = t; i < L; i += 256) mx = fmaxf(mx, pr[i]);
    red[t] = mx; __syncthreads();
    for (int st = 128; st > 0; st >>= 1) {
        if (t < st) red[t] = fmaxf(red[t], red[t + st]);
        __syncthreads();
    }
    mx = red[0]; __syncthreads();

    float sum = 0.f;
    for (int i = t; i < L; i += 256) {
        float e = __expf(pr[i] - mx);
        pr[i] = e; sum += e;
    }
    red[t] = sum; __syncthreads();
    for (int st = 128; st > 0; st >>= 1) {
        if (t < st) red[t] += red[t + st];
        __syncthreads();
    }
    const float inv = 1.f / red[0];
    for (int i = t; i < L; i += 256) pr[i] *= inv;
}

__global__ void colsum_kernel(const float* __restrict__ probs,
                              float* __restrict__ acc, int L, float w)
{
    const int s = blockIdx.x * blockDim.x + threadIdx.x;
    if (s >= L) return;
    float t = 0.f;
    for (int l = 0; l < L; ++l) t += probs[(size_t)l * L + s];
    acc[s] += t * w;
}

__global__ void zero_kernel(float* __restrict__ p, int n)
{
    const int i = blockIdx.x * blockDim.x + threadIdx.x;
    if (i < n) p[i] = 0.f;
}

__global__ void transpose_kernel(const float* __restrict__ in, int rows,
                                 int cols, int ldin, float* __restrict__ out)
{
    const int idx = blockIdx.x * blockDim.x + threadIdx.x;
    if (idx >= rows * cols) return;
    const int r = idx / cols, c = idx % cols;
    out[(size_t)c * rows + r] = in[(size_t)r * ldin + c];
}

__global__ void weighted_sum_kernel(const float* __restrict__ scores,
                                    const float* __restrict__ agg,
                                    float* __restrict__ emb)
{
    const int j = blockIdx.x * blockDim.x + threadIdx.x;
    if (j >= D_MODEL) return;
    float a = 0.f;
    for (int s = 0; s < N_SLICES; ++s) a += scores[s] * agg[(size_t)s * D_MODEL + j];
    emb[j] = a;
}

// ---------------------------------------------------------------------------
// Clinical MLP: relu(LN(W1 c + b1)) -> relu(LN(W2 . + b2)), single block.
// ---------------------------------------------------------------------------
__global__ void __launch_bounds__(512)
clinical_kernel(const float* __restrict__ clin,
                const float* __restrict__ w1, const float* __restrict__ b1,
                const float* __restrict__ g1, const float* __restrict__ bb1,
                const float* __restrict__ w2, const float* __restrict__ b2,
                const float* __restrict__ g2, const float* __restrict__ bb2,
                float* __restrict__ c2out)
{
    __shared__ float cl[CLIN_DIM], c1[CLIN_DIM], c2[D_MODEL], stat[2];
    const int t = threadIdx.x;
    if (t < CLIN_DIM) cl[t] = clin[t];
    __syncthreads();
    if (t < CLIN_DIM) {
        float a = b1[t];
        for (int j = 0; j < CLIN_DIM; ++j) a += w1[t * CLIN_DIM + j] * cl[j];
        c1[t] = a;
    }
    __syncthreads();
    if (t == 0) {
        float s = 0.f, s2 = 0.f;
        for (int i = 0; i < CLIN_DIM; ++i) { s += c1[i]; s2 += c1[i] * c1[i]; }
        float m = s / CLIN_DIM, v = s2 / CLIN_DIM - m * m;
        stat[0] = m; stat[1] = rsqrtf(v + 1e-5f);
    }
    __syncthreads();
    if (t < CLIN_DIM)
        c1[t] = fmaxf((c1[t] - stat[0]) * stat[1] * g1[t] + bb1[t], 0.f);
    __syncthreads();
    {
        float a = b2[t];
        for (int j = 0; j < CLIN_DIM; ++j) a += w2[t * CLIN_DIM + j] * c1[j];
        c2[t] = a;
    }
    __syncthreads();
    if (t == 0) {
        float s = 0.f, s2 = 0.f;
        for (int i = 0; i < D_MODEL; ++i) { s += c2[i]; s2 += c2[i] * c2[i]; }
        float m = s / D_MODEL, v = s2 / D_MODEL - m * m;
        stat[0] = m; stat[1] = rsqrtf(v + 1e-5f);
    }
    __syncthreads();
    c2out[t] = fmaxf((c2[t] - stat[0]) * stat[1] * g2[t] + bb2[t], 0.f);
}

// ---------------------------------------------------------------------------
// Cross-attention with a single KV token: softmax over one key == 1, so the
// per-position output is the constant vector out_w @ (Wv c + bv) + out_b.
// ---------------------------------------------------------------------------
__global__ void __launch_bounds__(512)
crossattn_vec_kernel(const float* __restrict__ in_w, const float* __restrict__ in_b,
                     const float* __restrict__ out_w, const float* __restrict__ out_b,
                     const float* __restrict__ c2, float* __restrict__ attv)
{
    __shared__ float cs[D_MODEL], V[D_MODEL];
    const int t = threadIdx.x;
    cs[t] = c2[t];
    __syncthreads();
    {   // V row: rows [2*D_MODEL, 3*D_MODEL) of in_w
        float a = in_b[2 * D_MODEL + t];
        const float* wr = in_w + (size_t)(2 * D_MODEL + t) * D_MODEL;
        for (int j = 0; j < D_MODEL; ++j) a += wr[j] * cs[j];
        V[t] = a;
    }
    __syncthreads();
    float o = out_b[t];
    const float* wo = out_w + (size_t)t * D_MODEL;
    for (int j = 0; j < D_MODEL; ++j) o += wo[j] * V[j];
    attv[t] = o;
}

// Final: out = relu(LN(op_w @ emb + op_b)), single 64-thread block.
__global__ void __launch_bounds__(64)
final_head_kernel(const float* __restrict__ op_w, const float* __restrict__ op_b,
                  const float* __restrict__ g, const float* __restrict__ b,
                  const float* __restrict__ emb, float* __restrict__ out)
{
    __shared__ float e[D_MODEL], h[64], stat[2];
    const int t = threadIdx.x;
    for (int i = t; i < D_MODEL; i += 64) e[i] = emb[i];
    __syncthreads();
    float a = op_b[t];
    const float* wr = op_w + (size_t)t * D_MODEL;
    for (int j = 0; j < D_MODEL; ++j) a += wr[j] * e[j];
    h[t] = a;
    __syncthreads();
    if (t == 0) {
        float s = 0.f, s2 = 0.f;
        for (int i = 0; i < 64; ++i) { s += h[i]; s2 += h[i] * h[i]; }
        float m = s / 64.f, v = s2 / 64.f - m * m;
        stat[0] = m; stat[1] = rsqrtf(v + 1e-5f);
    }
    __syncthreads();
    out[t] = fmaxf((h[t] - stat[0]) * stat[1] * g[t] + b[t], 0.f);
}

// ---------------------------------------------------------------------------
// Host-side orchestration
// ---------------------------------------------------------------------------
static inline void launch_gemm(const float* A, int lda, const float* B, int ldb,
                               float* C, int ldc, const float* bias,
                               int M, int N, int K, float scale, int act,
                               hipStream_t st)
{
    dim3 grid((N + 511) / 512, (M + 15) / 16);
    gemm_bf16_wmma<<<grid, 256, 0, st>>>(A, lda, B, ldb, C, ldc, bias,
                                         M, N, K, scale, act);
}

struct MambaW {
    const float *in_proj, *conv_w, *conv_b, *x_proj, *dt_w, *dt_b, *A_log, *Dp, *out_proj;
};

static void run_mamba(const float* x, float* out, const MambaW& w,
                      int numChunks, int Lc,
                      float* xz, float* xc, float* proj, float* dt, float* y,
                      hipStream_t st)
{
    // xz = x @ in_proj^T                     [2000, 2048]
    launch_gemm(x, D_MODEL, w.in_proj, D_MODEL, xz, 2 * D_INNER, nullptr,
                N_SLICES, 2 * D_INNER, D_MODEL, 1.f, ACT_NONE, st);
    // xc = silu(causal_conv(xr) + conv_b)    [2000, 1024]
    conv_silu_kernel<<<(N_SLICES * D_INNER + 255) / 256, 256, 0, st>>>(
        xz, w.conv_w, w.conv_b, xc, Lc);
    // proj = xc @ x_proj^T                   [2000, 64]
    launch_gemm(xc, D_INNER, w.x_proj, D_INNER, proj, DT_RANK + 2 * D_STATE,
                nullptr, N_SLICES, DT_RANK + 2 * D_STATE, D_INNER, 1.f, ACT_NONE, st);
    // dt = softplus(proj[:, :32] @ dt_w^T + dt_b)  [2000, 1024]
    launch_gemm(proj, DT_RANK + 2 * D_STATE, w.dt_w, DT_RANK, dt, D_INNER,
                w.dt_b, N_SLICES, D_INNER, DT_RANK, 1.f, ACT_SOFTPLUS, st);
    // selective scan + gate                  [2000, 1024]
    mamba_scan_kernel<<<numChunks * 4, 256, 0, st>>>(dt, xc, proj, xz,
                                                     w.A_log, w.Dp, y, Lc);
    // out = y @ out_proj^T                   [2000, 512]
    launch_gemm(y, D_INNER, w.out_proj, D_INNER, out, D_MODEL, nullptr,
                N_SLICES, D_MODEL, D_INNER, 1.f, ACT_NONE, st);
}

extern "C" void kernel_launch(void* const* d_in, const int* in_sizes, int n_in,
                              void* d_out, int out_size, void* d_ws, size_t ws_size,
                              hipStream_t stream)
{
    int a = 0;
    const float* slice_features = (const float*)d_in[a++];
    const float* clinical       = (const float*)d_in[a++];
    const float* enc_in_W  = (const float*)d_in[a++];
    const float* enc_in_b  = (const float*)d_in[a++];
    const float* enc_in_lg = (const float*)d_in[a++];
    const float* enc_in_lb = (const float*)d_in[a++];
    const float* em_in_proj = (const float*)d_in[a++];
    const float* em_conv_w  = (const float*)d_in[a++];
    const float* em_conv_b  = (const float*)d_in[a++];
    const float* em_x_proj  = (const float*)d_in[a++];
    const float* em_dt_w    = (const float*)d_in[a++];
    const float* em_dt_b    = (const float*)d_in[a++];
    const float* em_A_log   = (const float*)d_in[a++];
    const float* em_D       = (const float*)d_in[a++];
    const float* em_out_proj= (const float*)d_in[a++];
    const float* enc_ln_g   = (const float*)d_in[a++];
    const float* enc_ln_b   = (const float*)d_in[a++];
    const float* enc_out_W  = (const float*)d_in[a++];
    const float* enc_out_b  = (const float*)d_in[a++];
    const float* cl_w1 = (const float*)d_in[a++];
    const float* cl_b1 = (const float*)d_in[a++];
    const float* cl_l1g = (const float*)d_in[a++];
    const float* cl_l1b = (const float*)d_in[a++];
    const float* cl_w2 = (const float*)d_in[a++];
    const float* cl_b2 = (const float*)d_in[a++];
    const float* cl_l2g = (const float*)d_in[a++];
    const float* cl_l2b = (const float*)d_in[a++];
    const float* lm_in_proj = (const float*)d_in[a++];
    const float* lm_conv_w  = (const float*)d_in[a++];
    const float* lm_conv_b  = (const float*)d_in[a++];
    const float* lm_x_proj  = (const float*)d_in[a++];
    const float* lm_dt_w    = (const float*)d_in[a++];
    const float* lm_dt_b    = (const float*)d_in[a++];
    const float* lm_A_log   = (const float*)d_in[a++];
    const float* lm_D       = (const float*)d_in[a++];
    const float* lm_out_proj= (const float*)d_in[a++];
    const float* l_mln_g = (const float*)d_in[a++];
    const float* l_mln_b = (const float*)d_in[a++];
    const float* ca_in_w = (const float*)d_in[a++];
    const float* ca_in_b = (const float*)d_in[a++];
    const float* ca_out_w = (const float*)d_in[a++];
    const float* ca_out_b = (const float*)d_in[a++];
    const float* ca_ln_g = (const float*)d_in[a++];
    const float* ca_ln_b = (const float*)d_in[a++];
    const float* ff_w1 = (const float*)d_in[a++];
    const float* ff_b1 = (const float*)d_in[a++];
    const float* ff_w2 = (const float*)d_in[a++];
    const float* ff_b2 = (const float*)d_in[a++];
    const float* ffn_ln_g = (const float*)d_in[a++];
    const float* ffn_ln_b = (const float*)d_in[a++];
    const float* agg_in_w = (const float*)d_in[a++];
    const float* agg_in_b = (const float*)d_in[a++];
    const float* agg_out_w = (const float*)d_in[a++];
    const float* agg_out_b = (const float*)d_in[a++];
    const float* op_w = (const float*)d_in[a++];
    const float* op_b = (const float*)d_in[a++];
    const float* op_ln_g = (const float*)d_in[a++];
    const float* op_ln_b = (const float*)d_in[a++];

    // ---- workspace layout (floats) ----
    float* ws = (float*)d_ws;
    float* buf_h    = ws;                       // [2000,512]   sequence / encoder h
    float* buf_m    = buf_h  + 1024000;         // [2000,512]   mamba/ffn output
    float* buf_xz   = buf_m  + 1024000;         // [2000,2048]  (alias: attn probs)
    float* buf_xc   = buf_xz + 4096000;         // [2000,1024]  (alias: attn o)
    float* buf_dt   = buf_xc + 2048000;         // [2000,1024]  (alias: agg out)
    float* buf_y    = buf_dt + 2048000;         // [2000,1024]  (alias: V^T)
    float* buf_proj = buf_y  + 2048000;         // [2000,64]
    float* buf_qkv  = buf_proj + 128000;        // [2000,1536]
    float* buf_c2   = buf_qkv + 3072000;        // [512]
    float* buf_attv = buf_c2 + 512;             // [512]
    float* buf_emb  = buf_attv + 512;           // [512]
    float* buf_cols = buf_emb + 512;            // [2000]

    // ================= Encoder =================
    launch_gemm(slice_features, MRI_DIM, enc_in_W, MRI_DIM, buf_h, D_MODEL,
                enc_in_b, N_SLICES, D_MODEL, MRI_DIM, 1.f, ACT_NONE, stream);
    ln_kernel<<<N_SLICES, 256, 0, stream>>>(buf_h, nullptr, 0, enc_in_lg,
                                            enc_in_lb, buf_h, D_MODEL, 0);
    for (int i = 0; i < 2; ++i) {
        MambaW w = {
            em_in_proj + (size_t)i * 2 * D_INNER * D_MODEL,
            em_conv_w  + (size_t)i * D_INNER * D_CONV,
            em_conv_b  + (size_t)i * D_INNER,
            em_x_proj  + (size_t)i * (DT_RANK + 2 * D_STATE) * D_INNER,
            em_dt_w    + (size_t)i * D_INNER * DT_RANK,
            em_dt_b    + (size_t)i * D_INNER,
            em_A_log   + (size_t)i * D_INNER * D_STATE,
            em_D       + (size_t)i * D_INNER,
            em_out_proj+ (size_t)i * D_MODEL * D_INNER };
        run_mamba(buf_h, buf_m, w, N_SLICES / CHUNK, CHUNK,
                  buf_xz, buf_xc, buf_proj, buf_dt, buf_y, stream);
        ln_kernel<<<N_SLICES, 256, 0, stream>>>(buf_m, buf_h, D_MODEL,
            enc_ln_g + i * D_MODEL, enc_ln_b + i * D_MODEL, buf_h, D_MODEL, 0);
    }
    launch_gemm(buf_h, D_MODEL, enc_out_W, D_MODEL, buf_m, D_MODEL, enc_out_b,
                N_SLICES, D_MODEL, D_MODEL, 1.f, ACT_NONE, stream);
    hipMemcpyAsync(buf_h, buf_m, (size_t)N_SLICES * D_MODEL * sizeof(float),
                   hipMemcpyDeviceToDevice, stream);

    // ================= Clinical branch =================
    clinical_kernel<<<1, 512, 0, stream>>>(clinical, cl_w1, cl_b1, cl_l1g, cl_l1b,
                                           cl_w2, cl_b2, cl_l2g, cl_l2b, buf_c2);

    // ================= Transformer-Mamba layers =================
    for (int i = 0; i < N_LAYERS; ++i) {
        MambaW w = {
            lm_in_proj + (size_t)i * 2 * D_INNER * D_MODEL,
            lm_conv_w  + (size_t)i * D_INNER * D_CONV,
            lm_conv_b  + (size_t)i * D_INNER,
            lm_x_proj  + (size_t)i * (DT_RANK + 2 * D_STATE) * D_INNER,
            lm_dt_w    + (size_t)i * D_INNER * DT_RANK,
            lm_dt_b    + (size_t)i * D_INNER,
            lm_A_log   + (size_t)i * D_INNER * D_STATE,
            lm_D       + (size_t)i * D_INNER,
            lm_out_proj+ (size_t)i * D_MODEL * D_INNER };
        run_mamba(buf_h, buf_m, w, 1, N_SLICES,
                  buf_xz, buf_xc, buf_proj, buf_dt, buf_y, stream);
        ln_kernel<<<N_SLICES, 256, 0, stream>>>(buf_m, buf_h, D_MODEL,
            l_mln_g + i * D_MODEL, l_mln_b + i * D_MODEL, buf_h, D_MODEL, 0);

        // cross-attention (1 KV token -> constant vector), then add+LN
        crossattn_vec_kernel<<<1, 512, 0, stream>>>(
            ca_in_w + (size_t)i * 3 * D_MODEL * D_MODEL,
            ca_in_b + (size_t)i * 3 * D_MODEL,
            ca_out_w + (size_t)i * D_MODEL * D_MODEL,
            ca_out_b + (size_t)i * D_MODEL, buf_c2, buf_attv);
        ln_kernel<<<N_SLICES, 256, 0, stream>>>(buf_h, buf_attv, 0,
            ca_ln_g + i * D_MODEL, ca_ln_b + i * D_MODEL, buf_h, D_MODEL, 0);

        // FFN: gelu(x W1^T + b1) W2^T + b2, add + LN
        launch_gemm(buf_h, D_MODEL, ff_w1 + (size_t)i * 2 * D_MODEL * D_MODEL,
                    D_MODEL, buf_y, 2 * D_MODEL, ff_b1 + (size_t)i * 2 * D_MODEL,
                    N_SLICES, 2 * D_MODEL, D_MODEL, 1.f, ACT_GELU, stream);
        launch_gemm(buf_y, 2 * D_MODEL, ff_w2 + (size_t)i * D_MODEL * 2 * D_MODEL,
                    2 * D_MODEL, buf_m, D_MODEL, ff_b2 + (size_t)i * D_MODEL,
                    N_SLICES, D_MODEL, 2 * D_MODEL, 1.f, ACT_NONE, stream);
        ln_kernel<<<N_SLICES, 256, 0, stream>>>(buf_m, buf_h, D_MODEL,
            ffn_ln_g + i * D_MODEL, ffn_ln_b + i * D_MODEL, buf_h, D_MODEL, 0);
    }

    // ================= Final self-attention aggregation =================
    launch_gemm(buf_h, D_MODEL, agg_in_w, D_MODEL, buf_qkv, 3 * D_MODEL,
                agg_in_b, N_SLICES, 3 * D_MODEL, D_MODEL, 1.f, ACT_NONE, stream);
    // V^T for the probs @ V GEMM (B operand must be [N,K] row-major)
    float* buf_vt = buf_y;                       // [512, 2000]
    transpose_kernel<<<(N_SLICES * D_MODEL + 255) / 256, 256, 0, stream>>>(
        buf_qkv + 2 * D_MODEL, N_SLICES, D_MODEL, 3 * D_MODEL, buf_vt);
    zero_kernel<<<(N_SLICES + 255) / 256, 256, 0, stream>>>(buf_cols, N_SLICES);

    float* probs = buf_xz;                       // [2000, 2000], reused per head
    float* buf_o = buf_xc;                       // [2000, 512]
    for (int h = 0; h < N_HEADS; ++h) {
        // s = (Q_h K_h^T) / sqrt(64)
        launch_gemm(buf_qkv + h * HEAD_DIM, 3 * D_MODEL,
                    buf_qkv + D_MODEL + h * HEAD_DIM, 3 * D_MODEL,
                    probs, N_SLICES, nullptr,
                    N_SLICES, N_SLICES, HEAD_DIM, 0.125f, ACT_NONE, stream);
        softmax_rows_kernel<<<N_SLICES, 256, 0, stream>>>(probs, N_SLICES);
        colsum_kernel<<<(N_SLICES + 255) / 256, 256, 0, stream>>>(
            probs, buf_cols, N_SLICES, 1.f / (float)(N_HEADS * N_SLICES));
        // o_h = probs @ V_h
        launch_gemm(probs, N_SLICES, buf_vt + (size_t)h * HEAD_DIM * N_SLICES,
                    N_SLICES, buf_o + h * HEAD_DIM, D_MODEL, nullptr,
                    N_SLICES, HEAD_DIM, N_SLICES, 1.f, ACT_NONE, stream);
    }
    float* buf_agg = buf_dt;                     // [2000, 512]
    launch_gemm(buf_o, D_MODEL, agg_out_w, D_MODEL, buf_agg, D_MODEL,
                agg_out_b, N_SLICES, D_MODEL, D_MODEL, 1.f, ACT_NONE, stream);
    weighted_sum_kernel<<<2, 256, 0, stream>>>(buf_cols, buf_agg, buf_emb);
    final_head_kernel<<<1, 64, 0, stream>>>(op_w, op_b, op_ln_g, op_ln_b,
                                            buf_emb, (float*)d_out);
}
